// TokenUOTBlock_39341900432010
// MI455X (gfx1250) — compile-verified, matrix operand order checked by reading, not run on hardware
//
#include <hip/hip_runtime.h>
#include <stdint.h>

typedef __attribute__((ext_vector_type(16))) _Float16 v16h;
typedef __attribute__((ext_vector_type(8)))  float    v8f;

#define NTOK   1024
#define DMODEL 512
#define NHEAD  16
#define DH     32
#define DFF    2048
#define KITER  20
#define NEG_INF (-__builtin_inff())

static __device__ __forceinline__ float sigmoidf_(float x) { return 1.0f / (1.0f + __expf(-x)); }

// combine two (max, sum) log-sum-exp partials
static __device__ __forceinline__ void lse_comb(float& m, float& s, float m2, float s2) {
  if (m2 == NEG_INF) return;
  if (m2 > m) { s = s * __expf(m - m2) + s2; m = m2; }
  else        { s += s2 * __expf(m2 - m); }
}

static __device__ __forceinline__ float wred32(float x) {
  #pragma unroll
  for (int o = 16; o > 0; o >>= 1) x += __shfl_xor(x, o, 32);
  return x;
}

// ---------------- weight convert + transpose: Wt[n][k] = (f16)W[k][n] ----------------
__global__ void k_cvt_wt(const float* __restrict__ W, _Float16* __restrict__ Wt, int K, int N) {
  int idx = blockIdx.x * 256 + threadIdx.x;
  if (idx >= N * K) return;
  int n = idx / K, k = idx - n * K;
  Wt[idx] = (_Float16)W[k * N + n];
}

// ---------------- LayerNorm over D=512 (optional residual-add), f16 out ----------------
__global__ void __launch_bounds__(256)
k_ln(const float* __restrict__ X, const float* __restrict__ resid,
     const float* __restrict__ maskv, const float* __restrict__ w,
     const float* __restrict__ b, float* __restrict__ h1out,
     _Float16* __restrict__ y16) {
  __shared__ float red[256];
  int n = blockIdx.x, t = threadIdx.x;
  float mk = maskv ? maskv[n] : 1.0f;
  float v0 = X[n * DMODEL + t], v1 = X[n * DMODEL + 256 + t];
  if (resid) { v0 += resid[n * DMODEL + t] * mk; v1 += resid[n * DMODEL + 256 + t] * mk; }
  if (h1out) { h1out[n * DMODEL + t] = v0; h1out[n * DMODEL + 256 + t] = v1; }
  red[t] = v0 + v1; __syncthreads();
  for (int o = 128; o > 0; o >>= 1) { if (t < o) red[t] += red[t + o]; __syncthreads(); }
  float mean = red[0] * (1.0f / DMODEL); __syncthreads();
  float d0 = v0 - mean, d1 = v1 - mean;
  red[t] = d0 * d0 + d1 * d1; __syncthreads();
  for (int o = 128; o > 0; o >>= 1) { if (t < o) red[t] += red[t + o]; __syncthreads(); }
  float rstd = rsqrtf(red[0] * (1.0f / DMODEL) + 1e-5f);
  y16[n * DMODEL + t]       = (_Float16)(d0 * rstd * w[t] + b[t]);
  y16[n * DMODEL + 256 + t] = (_Float16)(d1 * rstd * w[256 + t] + b[256 + t]);
}

// ---------------- generic WMMA f16 GEMM: C(MxN,f32) = A(MxK) * Bt(NxK)^T ----------------
// One wave computes one 16x16 tile; per-lane operands are contiguous 32B v16h loads.
// K-half striping across lane groups: half h holds k = 16h..16h+15 (contiguous), same for A and B.
__global__ void __launch_bounds__(256)
k_gemm(const _Float16* __restrict__ A, const _Float16* __restrict__ Bt,
       float* __restrict__ C, int M, int N, int K) {
  int wave = (blockIdx.x * 256 + threadIdx.x) >> 5;
  int lane = threadIdx.x & 31;
  int tilesN = N >> 4;
  int tm = (wave / tilesN) << 4;
  int tn = (wave % tilesN) << 4;
  if (tm >= M) return;                       // wave-uniform: EXEC stays all-ones for WMMA
  int koff = (lane >> 4) << 4;
  const _Float16* ap = A  + (size_t)(tm + (lane & 15)) * K + koff;
  const _Float16* bp = Bt + (size_t)(tn + (lane & 15)) * K + koff;
  v8f acc = {};
  for (int k = 0; k < K; k += 64) {
    __builtin_prefetch(ap + k + 256, 0, 1);
    __builtin_prefetch(bp + k + 256, 0, 1);
    v16h a0 = *(const v16h*)(ap + k);
    v16h b0 = *(const v16h*)(bp + k);
    v16h a1 = *(const v16h*)(ap + k + 32);
    v16h b1 = *(const v16h*)(bp + k + 32);
    acc = __builtin_amdgcn_wmma_f32_16x16x32_f16(false, a0, false, b0, (short)0, acc, false, false);
    acc = __builtin_amdgcn_wmma_f32_16x16x32_f16(false, a1, false, b1, (short)0, acc, false, false);
  }
  int row = tm + ((lane >> 4) << 3);
  int col = tn + (lane & 15);
  #pragma unroll
  for (int r = 0; r < 8; ++r) C[(size_t)(row + r) * N + col] = acc[r];
}

// ---------------- split QKVG, per-head LN on Q/K, V transposed, sigmoid(G) ----------------
__global__ void __launch_bounds__(256)
k_split(const float* __restrict__ QKVG, _Float16* __restrict__ Qh, _Float16* __restrict__ Kh,
        _Float16* __restrict__ Vt, float* __restrict__ SG) {
  int w = (blockIdx.x * 256 + threadIdx.x) >> 5;
  int d = threadIdx.x & 31;
  int n = w >> 4, hh = w & 15;
  const float* row = QKVG + (size_t)n * (4 * DMODEL) + hh * DH;
  float q = row[d], k = row[DMODEL + d], v = row[2 * DMODEL + d], g = row[3 * DMODEL + d];
  float qm = wred32(q) * (1.0f / 32.0f);
  float qd = q - qm;
  float qn = qd * rsqrtf(wred32(qd * qd) * (1.0f / 32.0f) + 1e-5f);
  float km = wred32(k) * (1.0f / 32.0f);
  float kd = k - km;
  float kn = kd * rsqrtf(wred32(kd * kd) * (1.0f / 32.0f) + 1e-5f);
  int base = (hh * NTOK + n) * DH + d;
  Qh[base] = (_Float16)qn;
  Kh[base] = (_Float16)kn;
  Vt[((hh * DH + d) << 10) + n] = (_Float16)v;
  SG[base] = sigmoidf_(g);
}

// ---------------- scores: M[h][i][j] = (QK/sqrt(dh) - pos_bias - wdist*fdist)/eps ----------------
__global__ void __launch_bounds__(256)
k_scores(const _Float16* __restrict__ Qh, const _Float16* __restrict__ Kh,
         const float* __restrict__ pos_bias, const float* __restrict__ xres,
         const float* __restrict__ epsv, const float* __restrict__ wdl,
         const float* __restrict__ geo, float* __restrict__ Mm) {
  int wave = (blockIdx.x * 256 + threadIdx.x) >> 5;
  int lane = threadIdx.x & 31;
  int tj = wave & 63, ti = (wave >> 6) & 63, hh = wave >> 12;
  int koff = (lane >> 4) << 4;
  v16h a = *(const v16h*)(Qh + (size_t)(hh * NTOK + ti * 16 + (lane & 15)) * DH + koff);
  v16h b = *(const v16h*)(Kh + (size_t)(hh * NTOK + tj * 16 + (lane & 15)) * DH + koff);
  v8f acc = {};
  acc = __builtin_amdgcn_wmma_f32_16x16x32_f16(false, a, false, b, (short)0, acc, false, false);
  float wdist = sigmoidf_(wdl[hh]) * geo[hh];
  float inv_eps = 1.0f / epsv[hh];
  const float isd = 0.17677669529663687f;  // 1/sqrt(32)
  int col = tj * 16 + (lane & 15);
  float xc0 = xres[col * 3 + 0], xc1 = xres[col * 3 + 1], xc2 = xres[col * 3 + 2];
  int rbase = ti * 16 + ((lane >> 4) << 3);
  #pragma unroll
  for (int r = 0; r < 8; ++r) {
    int row = rbase + r;
    float dx = xres[row * 3 + 0] - xc0, dy = xres[row * 3 + 1] - xc1, dz = xres[row * 3 + 2] - xc2;
    float dist = sqrtf(fmaxf(dx * dx + dy * dy + dz * dz, 1e-12f));
    float fd = dist / (10.0f + dist);
    float pb = pos_bias[(((size_t)hh << 10) + row) * NTOK + col];
    Mm[(((size_t)hh << 10) + row) * NTOK + col] = (acc[r] * isd - pb - wdist * fd) * inv_eps;
  }
}

// ---------------- copy two arrays (init / export log_u,log_v) ----------------
__global__ void k_copy2(const float* __restrict__ a, const float* __restrict__ b,
                        float* __restrict__ da, float* __restrict__ db, int n) {
  int i = blockIdx.x * 256 + threadIdx.x;
  if (i < n) { da[i] = a[i]; db[i] = b[i]; }
}

// ---------------- Sinkhorn row update: lu = fi*(log_mu - LSE_j(M + lv + pad_j)) ----------------
__global__ void __launch_bounds__(256)
k_row_lse(const float* __restrict__ Mm, const float* __restrict__ lv,
          const float* __restrict__ mu, const float* __restrict__ maskv,
          const float* __restrict__ epsv, float* __restrict__ lu) {
  __shared__ float sm[256], ss[256];
  int hh = blockIdx.x >> 10, i = blockIdx.x & 1023, t = threadIdx.x;
  const float* mrow = Mm + ((size_t)(hh * NTOK + i) << 10);
  const float* lvh = lv + hh * NTOK;
  float m = NEG_INF, s = 0.0f;
  for (int j = t; j < NTOK; j += 256) {
    float pad = maskv[j] > 0.0f ? 0.0f : -1e9f;
    lse_comb(m, s, mrow[j] + lvh[j] + pad, 1.0f);
  }
  sm[t] = m; ss[t] = s; __syncthreads();
  for (int o = 128; o > 0; o >>= 1) {
    if (t < o) { float m2 = sm[t + o], s2 = ss[t + o]; lse_comb(sm[t], ss[t], m2, s2); }
    __syncthreads();
  }
  if (t == 0) {
    float lse = (sm[0] == NEG_INF) ? -1e30f : sm[0] + __logf(ss[0]);
    float fi = 1.0f / (1.0f + epsv[hh]);  // LAM = 1
    lu[hh * NTOK + i] = fi * (__logf(fmaxf(mu[hh * NTOK + i], 1e-8f)) - lse);
  }
}

// ---------------- Sinkhorn column update, phase 1: partial LSE over 128-row strips ----------------
__global__ void __launch_bounds__(256)
k_colpart(const float* __restrict__ Mm, const float* __restrict__ lu,
          const float* __restrict__ maskv, float* __restrict__ pm, float* __restrict__ ps) {
  int b = blockIdx.x;                  // 16h x 8rb x 4cb = 512 blocks
  int cb = b & 3, rb = (b >> 2) & 7, hh = b >> 5;
  int col = cb * 256 + threadIdx.x;
  const float* luh = lu + hh * NTOK;
  float m = NEG_INF, s = 0.0f;
  int i0 = rb * 128;
  for (int i = i0; i < i0 + 128; ++i) {
    float pad = maskv[i] > 0.0f ? 0.0f : -1e9f;
    float x = Mm[((size_t)(hh * NTOK + i) << 10) + col] + luh[i] + pad;
    lse_comb(m, s, x, 1.0f);
  }
  pm[(hh * 8 + rb) * NTOK + col] = m;
  ps[(hh * 8 + rb) * NTOK + col] = s;
}

// ---------------- Sinkhorn column update, phase 2: combine strips, write lv ----------------
__global__ void __launch_bounds__(256)
k_colfin(const float* __restrict__ pm, const float* __restrict__ ps,
         const float* __restrict__ nu, const float* __restrict__ epsv,
         float* __restrict__ lv) {
  int b = blockIdx.x;                  // 16h x 4cb = 64 blocks
  int cb = b & 3, hh = b >> 2;
  int col = cb * 256 + threadIdx.x;
  float m = NEG_INF, s = 0.0f;
  #pragma unroll
  for (int rb = 0; rb < 8; ++rb)
    lse_comb(m, s, pm[(hh * 8 + rb) * NTOK + col], ps[(hh * 8 + rb) * NTOK + col]);
  float lse = (m == NEG_INF) ? -1e30f : m + __logf(s);
  float fi = 1.0f / (1.0f + epsv[hh]);
  lv[hh * NTOK + col] = fi * (__logf(fmaxf(nu[hh * NTOK + col], 1e-8f)) - lse);
}

// ---------------- row sums of P = exp(M + lu_i + lv_j + pad_j); invS = 1/(S+1e-9) ----------------
__global__ void __launch_bounds__(256)
k_rowsum(const float* __restrict__ Mm, const float* __restrict__ lu,
         const float* __restrict__ lv, const float* __restrict__ maskv,
         float* __restrict__ invS) {
  __shared__ float red[256];
  int hh = blockIdx.x >> 10, i = blockIdx.x & 1023, t = threadIdx.x;
  const float* mrow = Mm + ((size_t)(hh * NTOK + i) << 10);
  const float* lvh = lv + hh * NTOK;
  float lui = lu[hh * NTOK + i];
  float s = 0.0f;
  for (int j = t; j < NTOK; j += 256) {
    float pad = maskv[j] > 0.0f ? 0.0f : -1e9f;
    s += __expf(mrow[j] + lui + lvh[j] + pad);
  }
  red[t] = s; __syncthreads();
  for (int o = 128; o > 0; o >>= 1) { if (t < o) red[t] += red[t + o]; __syncthreads(); }
  if (t == 0) invS[hh * NTOK + i] = 1.0f / (red[0] + 1e-9f);
}

// ---------------- o = (Pn @ V) * sigmoid(G) via WMMA, f16 out in [n][h*32+d] layout ----------------
__global__ void __launch_bounds__(256)
k_attnV(const float* __restrict__ Mm, const float* __restrict__ lu,
        const float* __restrict__ lv, const float* __restrict__ maskv,
        const float* __restrict__ invS, const _Float16* __restrict__ Vt,
        const float* __restrict__ SG, _Float16* __restrict__ o16) {
  int wave = (blockIdx.x * 256 + threadIdx.x) >> 5;
  int lane = threadIdx.x & 31;
  int hh = wave >> 6, ti = wave & 63;
  int koff = (lane >> 4) << 4;
  int arow = ti * 16 + (lane & 15);
  float lui = lu[hh * NTOK + arow];
  float isr = invS[hh * NTOK + arow];
  const float* Mr = Mm + ((size_t)(hh * NTOK + arow) << 10) + koff;
  const float* lvh = lv + hh * NTOK;
  const _Float16* v0p = Vt + ((size_t)(hh * DH + (lane & 15)) << 10) + koff;
  const _Float16* v1p = Vt + ((size_t)(hh * DH + 16 + (lane & 15)) << 10) + koff;
  v8f c0 = {}, c1 = {};
  for (int jc = 0; jc < NTOK; jc += 32) {
    v16h a;
    #pragma unroll
    for (int tt = 0; tt < 16; ++tt) {
      int j = jc + koff + tt;
      float pad = maskv[j] > 0.0f ? 0.0f : -1e9f;
      a[tt] = (_Float16)(__expf(Mr[jc + tt] + lui + lvh[j] + pad) * isr);
    }
    v16h b0 = *(const v16h*)(v0p + jc);
    v16h b1 = *(const v16h*)(v1p + jc);
    c0 = __builtin_amdgcn_wmma_f32_16x16x32_f16(false, a, false, b0, (short)0, c0, false, false);
    c1 = __builtin_amdgcn_wmma_f32_16x16x32_f16(false, a, false, b1, (short)0, c1, false, false);
  }
  int rbase = ti * 16 + ((lane >> 4) << 3);
  int d0 = lane & 15, d1 = 16 + (lane & 15);
  #pragma unroll
  for (int r = 0; r < 8; ++r) {
    int row = rbase + r;
    o16[(size_t)row * DMODEL + hh * DH + d0] =
        (_Float16)(c0[r] * SG[(hh * NTOK + row) * DH + d0]);
    o16[(size_t)row * DMODEL + hh * DH + d1] =
        (_Float16)(c1[r] * SG[(hh * NTOK + row) * DH + d1]);
  }
}

// ---------------- SwiGLU gate: g = silu(u) * v ----------------
__global__ void k_gate(const float* __restrict__ U, _Float16* __restrict__ G) {
  int idx = blockIdx.x * 256 + threadIdx.x;
  if (idx >= NTOK * DFF) return;
  int n = idx >> 11, c = idx & 2047;
  float u = U[(size_t)n * (2 * DFF) + c];
  float v = U[(size_t)n * (2 * DFF) + DFF + c];
  G[idx] = (_Float16)(u * sigmoidf_(u) * v);
}

// ---------------- centroids: xc[h][i][:] = Pn @ x_res ----------------
__global__ void __launch_bounds__(256)
k_centroid(const float* __restrict__ Mm, const float* __restrict__ lu,
           const float* __restrict__ lv, const float* __restrict__ maskv,
           const float* __restrict__ invS, const float* __restrict__ xres,
           float* __restrict__ xc) {
  __shared__ float s0[256], s1[256], s2[256];
  int hh = blockIdx.x >> 10, i = blockIdx.x & 1023, t = threadIdx.x;
  const float* mrow = Mm + ((size_t)(hh * NTOK + i) << 10);
  const float* lvh = lv + hh * NTOK;
  float lui = lu[hh * NTOK + i], isr = invS[hh * NTOK + i];
  float a0 = 0, a1 = 0, a2 = 0;
  for (int j = t; j < NTOK; j += 256) {
    float pad = maskv[j] > 0.0f ? 0.0f : -1e9f;
    float p = __expf(mrow[j] + lui + lvh[j] + pad) * isr;
    a0 += p * xres[j * 3 + 0]; a1 += p * xres[j * 3 + 1]; a2 += p * xres[j * 3 + 2];
  }
  s0[t] = a0; s1[t] = a1; s2[t] = a2; __syncthreads();
  for (int o = 128; o > 0; o >>= 1) {
    if (t < o) { s0[t] += s0[t + o]; s1[t] += s1[t + o]; s2[t] += s2[t + o]; }
    __syncthreads();
  }
  if (t == 0) {
    xc[(hh * NTOK + i) * 3 + 0] = s0[0];
    xc[(hh * NTOK + i) * 3 + 1] = s1[0];
    xc[(hh * NTOK + i) * 3 + 2] = s2[0];
  }
}

// ---------------- final h output: h1 + ff*mask ----------------
__global__ void k_out_h(const float* __restrict__ h1, const float* __restrict__ ff,
                        const float* __restrict__ maskv, float* __restrict__ out) {
  int idx = blockIdx.x * 256 + threadIdx.x;
  if (idx >= NTOK * DMODEL) return;
  out[idx] = h1[idx] + ff[idx] * maskv[idx >> 9];
}

// ---------------- x_res output ----------------
__global__ void k_out_x(const float* __restrict__ xres, const float* __restrict__ xc,
                        const float* __restrict__ gamma, const float* __restrict__ maskv,
                        float* __restrict__ out) {
  int idx = blockIdx.x * 256 + threadIdx.x;
  if (idx >= NTOK * 3) return;
  int i = idx / 3, c = idx - i * 3;
  float x = xres[idx];
  float acc = 0.0f;
  #pragma unroll
  for (int hh = 0; hh < NHEAD; ++hh)
    acc += tanhf(gamma[hh]) * (x - xc[(hh * NTOK + i) * 3 + c]);
  out[idx] = x + acc * maskv[i];
}

extern "C" void kernel_launch(void* const* d_in, const int* in_sizes, int n_in,
                              void* d_out, int out_size, void* d_ws, size_t ws_size,
                              hipStream_t stream) {
  (void)in_sizes; (void)n_in; (void)out_size; (void)ws_size;
  const float* h    = (const float*)d_in[0];
  const float* xres = (const float*)d_in[1];
  const float* mu   = (const float*)d_in[2];
  const float* nu   = (const float*)d_in[3];
  const float* lu0  = (const float*)d_in[4];
  const float* lv0  = (const float*)d_in[5];
  const float* pb   = (const float*)d_in[6];
  /* d_in[7] = pos_bins (unused by reference) */
  const float* geo  = (const float*)d_in[8];
  const float* mask = (const float*)d_in[9];
  const float* lnaw = (const float*)d_in[10];
  const float* lnab = (const float*)d_in[11];
  const float* wq   = (const float*)d_in[12];
  const float* wk   = (const float*)d_in[13];
  const float* wv   = (const float*)d_in[14];
  const float* wg   = (const float*)d_in[15];
  const float* wo   = (const float*)d_in[16];
  const float* gam  = (const float*)d_in[17];
  const float* wdl  = (const float*)d_in[18];
  const float* epsv = (const float*)d_in[19];
  const float* lnfw = (const float*)d_in[20];
  const float* lnfb = (const float*)d_in[21];
  const float* w1   = (const float*)d_in[22];
  const float* w2   = (const float*)d_in[23];
  const float* w3   = (const float*)d_in[24];
  float* out = (float*)d_out;

  uint8_t* p = (uint8_t*)d_ws;
  auto alloc = [&](size_t bytes) -> void* {
    void* r = (void*)p; p += (bytes + 255) & ~(size_t)255; return r;
  };
  _Float16* hn16    = (_Float16*)alloc((size_t)NTOK * DMODEL * 2);
  _Float16* wqkvg_t = (_Float16*)alloc((size_t)2048 * 512 * 2);
  _Float16* wo_t    = (_Float16*)alloc((size_t)512 * 512 * 2);
  _Float16* w12_t   = (_Float16*)alloc((size_t)4096 * 512 * 2);
  _Float16* w3_t    = (_Float16*)alloc((size_t)512 * 2048 * 2);
  float*    QKVG    = (float*)   alloc((size_t)NTOK * 2048 * 4);
  _Float16* Qh      = (_Float16*)alloc((size_t)NHEAD * NTOK * DH * 2);
  _Float16* Kh      = (_Float16*)alloc((size_t)NHEAD * NTOK * DH * 2);
  _Float16* Vt      = (_Float16*)alloc((size_t)NHEAD * DH * NTOK * 2);
  float*    SG      = (float*)   alloc((size_t)NHEAD * NTOK * DH * 4);
  float*    Mm      = (float*)   alloc((size_t)NHEAD * NTOK * NTOK * 4);   // 64 MB, L2-resident
  float*    lu      = (float*)   alloc((size_t)NHEAD * NTOK * 4);
  float*    lv      = (float*)   alloc((size_t)NHEAD * NTOK * 4);
  float*    pm      = (float*)   alloc((size_t)NHEAD * 8 * NTOK * 4);
  float*    ps      = (float*)   alloc((size_t)NHEAD * 8 * NTOK * 4);
  float*    invS    = (float*)   alloc((size_t)NHEAD * NTOK * 4);
  _Float16* o16     = (_Float16*)alloc((size_t)NTOK * DMODEL * 2);
  float*    OWO     = (float*)   alloc((size_t)NTOK * DMODEL * 4);
  float*    h1      = (float*)   alloc((size_t)NTOK * DMODEL * 4);
  _Float16* hf16    = (_Float16*)alloc((size_t)NTOK * DMODEL * 2);
  float*    U       = (float*)   alloc((size_t)NTOK * 4096 * 4);
  _Float16* Gff     = (_Float16*)alloc((size_t)NTOK * DFF * 2);
  float*    FF      = (float*)   alloc((size_t)NTOK * DMODEL * 4);
  float*    xc      = (float*)   alloc((size_t)NHEAD * NTOK * 3 * 4);

  // 1) weight f32->f16 transposed (Bt layout for WMMA B operand)
  k_cvt_wt<<<1024, 256, 0, stream>>>(wq, wqkvg_t + (size_t)0 * 512 * 512, 512, 512);
  k_cvt_wt<<<1024, 256, 0, stream>>>(wk, wqkvg_t + (size_t)1 * 512 * 512, 512, 512);
  k_cvt_wt<<<1024, 256, 0, stream>>>(wv, wqkvg_t + (size_t)2 * 512 * 512, 512, 512);
  k_cvt_wt<<<1024, 256, 0, stream>>>(wg, wqkvg_t + (size_t)3 * 512 * 512, 512, 512);
  k_cvt_wt<<<1024, 256, 0, stream>>>(wo, wo_t, 512, 512);
  k_cvt_wt<<<4096, 256, 0, stream>>>(w1, w12_t, 512, 2048);
  k_cvt_wt<<<4096, 256, 0, stream>>>(w2, w12_t + (size_t)2048 * 512, 512, 2048);
  k_cvt_wt<<<4096, 256, 0, stream>>>(w3, w3_t, 2048, 512);

  // 2) LN(h) -> f16
  k_ln<<<NTOK, 256, 0, stream>>>(h, nullptr, nullptr, lnaw, lnab, nullptr, hn16);

  // 3) fused QKVG projection: 1024x2048x512 WMMA GEMM
  k_gemm<<<(64 * 128) / 8, 256, 0, stream>>>(hn16, wqkvg_t, QKVG, NTOK, 2048, DMODEL);

  // 4) per-head LN(Q), LN(K), V transpose, sigmoid(G)
  k_split<<<2048, 256, 0, stream>>>(QKVG, Qh, Kh, Vt, SG);

  // 5) scores M (WMMA QK^T + bias + geometric term)
  k_scores<<<8192, 256, 0, stream>>>(Qh, Kh, pb, xres, epsv, wdl, geo, Mm);

  // 6) Sinkhorn init + 20 iterations (M stays in L2: 64MB < 192MB)
  k_copy2<<<64, 256, 0, stream>>>(lu0, lv0, lu, lv, NHEAD * NTOK);
  for (int it = 0; it < KITER; ++it) {
    k_row_lse<<<NHEAD * NTOK, 256, 0, stream>>>(Mm, lv, mu, mask, epsv, lu);
    k_colpart<<<512, 256, 0, stream>>>(Mm, lu, mask, pm, ps);
    k_colfin<<<64, 256, 0, stream>>>(pm, ps, nu, epsv, lv);
  }

  // 7) transport-plan row sums, then Pn@V (WMMA) and centroids
  k_rowsum<<<NHEAD * NTOK, 256, 0, stream>>>(Mm, lu, lv, mask, invS);
  k_attnV<<<(NHEAD * 64) / 8, 256, 0, stream>>>(Mm, lu, lv, mask, invS, Vt, SG, o16);
  k_centroid<<<NHEAD * NTOK, 256, 0, stream>>>(Mm, lu, lv, mask, invS, xres, xc);

  // 8) output projection + residual + FFN LN
  k_gemm<<<(64 * 32) / 8, 256, 0, stream>>>(o16, wo_t, OWO, NTOK, DMODEL, DMODEL);
  k_ln<<<NTOK, 256, 0, stream>>>(h, OWO, mask, lnfw, lnfb, h1, hf16);

  // 9) SwiGLU FFN
  k_gemm<<<(64 * 256) / 8, 256, 0, stream>>>(hf16, w12_t, U, NTOK, 4096, DMODEL);
  k_gate<<<(NTOK * DFF) / 256, 256, 0, stream>>>(U, Gff);
  k_gemm<<<(64 * 32) / 8, 256, 0, stream>>>(Gff, w3_t, FF, NTOK, DMODEL, DFF);

  // 10) outputs: h (524288) | x_res (3072) | log_u (16384) | log_v (16384)
  k_out_h<<<(NTOK * DMODEL) / 256, 256, 0, stream>>>(h1, FF, mask, out);
  k_out_x<<<12, 256, 0, stream>>>(xres, xc, gam, mask, out + 524288);
  k_copy2<<<64, 256, 0, stream>>>(lu, lv, out + 527360, out + 543744, NHEAD * NTOK);
}
